// EdgeMoEGater_88742614270593
// MI455X (gfx1250) — matrix operand order repaired and checked by reading, not compiled, not run on hardware
//
#include <hip/hip_runtime.h>
#include <hip/hip_bf16.h>

typedef __attribute__((ext_vector_type(16))) __bf16 v16bf;
typedef __attribute__((ext_vector_type(8)))  float  v8f;
typedef __attribute__((ext_vector_type(4)))  float  v4f;
typedef __attribute__((ext_vector_type(4)))  int    v4i;

#define E_EDGES 640000
#define DFEAT   128
#define HID     64
#define NEXP    8
#define FRAG_ELEMS 512        // 32 lanes * 16 bf16 per B fragment
#define NFRAGS  22            // 16 (w1) + 2 (w2) + 4 (wproj)
#define TILES_PER_WAVE 4      // 4 waves * 4 tiles * 16 edges = 256 edges/block

#define V8F_ZERO ((v8f){0.f,0.f,0.f,0.f,0.f,0.f,0.f,0.f})

#if defined(__has_builtin)
#if __has_builtin(__builtin_amdgcn_global_load_async_to_lds_b128)
#define HAVE_ASYNC_LDS 1
#endif
#endif

// address-space qualified int4 for the async-copy builtin
typedef __attribute__((address_space(1))) v4i gv4i;   // global
typedef __attribute__((address_space(3))) v4i lv4i;   // LDS

// ---------------------------------------------------------------------------
// Workspace layout (bf16): prepacked WMMA B-operand fragments.
//   frags  0..15 : w1    (c = K-chunk 0..3) * 4 + (t = N-tile 0..3)
//   frags 16..17 : w2    (c = K-chunk 0..1), N padded 8->16 with zeros
//   frags 18..21 : wproj (c = K-chunk 0..3), N padded 8->16 with zeros
// Fragment element j of lane l  <-  B[k0 + (l>>4)*8 + (j>>3)*16 + (j&7)][n0 + (l&15)]
// ---------------------------------------------------------------------------
__global__ void pack_weights(const float* __restrict__ w1,
                             const float* __restrict__ w2,
                             const float* __restrict__ wp,
                             __bf16* __restrict__ wsb) {
  const int f    = blockIdx.x;       // fragment id 0..21
  const int i    = threadIdx.x;      // 0..511
  const int lane = i >> 4;           // 0..31
  const int j    = i & 15;           // 0..15
  const int kl   = ((lane >> 4) << 3) + ((j >> 3) << 4) + (j & 7);  // k within 32-chunk
  const int n    = lane & 15;

  float v = 0.f;
  if (f < 16) {                       // w1: [128 x 64]
    const int c = f >> 2, t = f & 3;
    v = w1[(32 * c + kl) * HID + (16 * t + n)];
  } else if (f < 18) {                // w2: [64 x 8], pad N to 16
    const int c = f - 16;
    if (n < NEXP) v = w2[(32 * c + kl) * NEXP + n];
  } else {                            // wproj: [128 x 8], pad N to 16
    const int c = f - 18;
    if (n < NEXP) v = wp[(32 * c + kl) * NEXP + n];
  }
  wsb[(size_t)f * FRAG_ELEMS + lane * 16 + j] = (__bf16)v;
}

// Pack four float4s (raw row data) into one 16x32 bf16 A fragment.
__device__ __forceinline__ v16bf cvt_frag(v4f f0, v4f f1, v4f f2, v4f f3) {
  v16bf a;
  a[0]  = (__bf16)f0[0]; a[1]  = (__bf16)f0[1]; a[2]  = (__bf16)f0[2]; a[3]  = (__bf16)f0[3];
  a[4]  = (__bf16)f1[0]; a[5]  = (__bf16)f1[1]; a[6]  = (__bf16)f1[2]; a[7]  = (__bf16)f1[3];
  a[8]  = (__bf16)f2[0]; a[9]  = (__bf16)f2[1]; a[10] = (__bf16)f2[2]; a[11] = (__bf16)f2[3];
  a[12] = (__bf16)f3[0]; a[13] = (__bf16)f3[1]; a[14] = (__bf16)f3[2]; a[15] = (__bf16)f3[3];
  return a;
}

// Load TWO consecutive 32-wide K-chunks of a row as A fragments.  All eight
// 16B loads are issued before any conversion so they clause with one wait.
// Element j of chunk c holds row[32c + (l>>4)*8 + (j>>3)*16 + (j&7)].
__device__ __forceinline__ void load_a_pair(const float* __restrict__ row,
                                            int kb0, v16bf& a0, v16bf& a1) {
  const int kb1 = kb0 + 32;
  v4f r0 = *(const v4f*)(row + kb0);
  v4f r1 = *(const v4f*)(row + kb0 + 4);
  v4f r2 = *(const v4f*)(row + kb0 + 16);
  v4f r3 = *(const v4f*)(row + kb0 + 20);
  v4f r4 = *(const v4f*)(row + kb1);
  v4f r5 = *(const v4f*)(row + kb1 + 4);
  v4f r6 = *(const v4f*)(row + kb1 + 16);
  v4f r7 = *(const v4f*)(row + kb1 + 20);
  a0 = cvt_frag(r0, r1, r2, r3);
  a1 = cvt_frag(r4, r5, r6, r7);
}

// ---------------------------------------------------------------------------
// Main fused kernel: 4 waves / block, 4 x 16-edge tiles per wave.
// All B fragments staged once per block into LDS via async global->LDS copies.
// ---------------------------------------------------------------------------
__global__ __launch_bounds__(128) void moe_fused(
    const float* __restrict__ x,   const float* __restrict__ b1,
    const float* __restrict__ b2,  const __bf16* __restrict__ wsb,
    float* __restrict__ out_fused, float* __restrict__ out_alpha) {
  __shared__ __align__(32) __bf16 ldsB[NFRAGS * FRAG_ELEMS];  // 22528 B
  __shared__ float lds_h[4][16][68];                           // 17408 B

  const int tid  = threadIdx.x;
  const int wv   = tid >> 5;
  const int lane = tid & 31;
  const int mrow = lane & 15;
  const int hh   = lane >> 4;

  // ---- stage packed B fragments into LDS (22528 B = 1408 x 16B chunks) ----
#if HAVE_ASYNC_LDS
  {
    gv4i* src = (gv4i*)wsb;
    lv4i* dst = (lv4i*)&ldsB[0];
#pragma unroll
    for (int k = 0; k < 11; ++k) {
      const int idx = tid + 128 * k;        // 16B chunk index
      __builtin_amdgcn_global_load_async_to_lds_b128(src + idx, dst + idx, 0, 0);
    }
#if defined(__has_builtin) && __has_builtin(__builtin_amdgcn_s_wait_asynccnt)
    __builtin_amdgcn_s_wait_asynccnt(0);
#else
    asm volatile("s_wait_asynccnt 0x0" ::: "memory");
#endif
  }
#else
#pragma unroll
  for (int k = 0; k < 11; ++k) {
    const int off = (tid + 128 * k) * 16;
    *(v4f*)((char*)ldsB + off) = *(const v4f*)((const char*)wsb + off);
  }
#endif
  __syncthreads();

#define BFRAG(f) (*(const v16bf*)(ldsB + (f) * FRAG_ELEMS + lane * 16))

  for (int it = 0; it < TILES_PER_WAVE; ++it) {
    const int e0 = blockIdx.x * (64 * TILES_PER_WAVE) +
                   wv * (16 * TILES_PER_WAVE) + it * 16;
    const float* xr = x + (size_t)(e0 + mrow) * DFEAT;

    // speculative prefetch of next tile's row (dropped if OOB)
    if (it + 1 < TILES_PER_WAVE) __builtin_prefetch(xr + 16 * DFEAT, 0, 0);

    // ---- A fragments of x (shared by x@w1 and x@w_proj) ----
    v16bf a[4];
    load_a_pair(xr, hh * 8,      a[0], a[1]);   // chunks 0,1
    load_a_pair(xr, 64 + hh * 8, a[2], a[3]);   // chunks 2,3

    // ---- h = relu(x @ w1 + b1): 4 N-tiles x 4 K-chunks = 16 WMMAs ----
#pragma unroll
    for (int t = 0; t < 4; ++t) {
      v8f acc = V8F_ZERO;
#pragma unroll
      for (int c = 0; c < 4; ++c) {
        v16bf bw = BFRAG(c * 4 + t);
        acc = __builtin_amdgcn_wmma_f32_16x16x32_bf16(false, a[c], false, bw,
                                                      (short)0, acc, false, false);
      }
      const float bv = b1[t * 16 + mrow];          // column n = 16t + (lane&15)
#pragma unroll
      for (int r = 0; r < 8; ++r)                  // row M = r + 8*(lane>>4)
        lds_h[wv][r + 8 * hh][t * 16 + mrow] = fmaxf(acc[r] + bv, 0.f);
    }

    // ---- expert_scores = x @ w_proj: 4 WMMAs (N padded to 16) ----
    v8f accp = V8F_ZERO;
#pragma unroll
    for (int c = 0; c < 4; ++c) {
      v16bf bw = BFRAG(18 + c);
      accp = __builtin_amdgcn_wmma_f32_16x16x32_bf16(false, a[c], false, bw,
                                                     (short)0, accp, false, false);
    }

    // ---- logits = h @ w2: reload h from LDS in A layout, 2 WMMAs ----
    const float* hr = &lds_h[wv][mrow][0];         // DS ops in-order within wave
    v16bf ah0, ah1;
    load_a_pair(hr, hh * 8, ah0, ah1);             // chunks 0,1 of K=64
    v8f accl = V8F_ZERO;
    accl = __builtin_amdgcn_wmma_f32_16x16x32_bf16(false, ah0, false, BFRAG(16),
                                                   (short)0, accl, false, false);
    accl = __builtin_amdgcn_wmma_f32_16x16x32_bf16(false, ah1, false, BFRAG(17),
                                                   (short)0, accl, false, false);

    // padded expert columns (n >= 8) get -1e30 -> softmax weight exactly 0
    const float b2v = (mrow < NEXP) ? b2[mrow] : -1e30f;

    // ---- softmax over K=8 + alpha-weighted reduce, within 16-lane groups ----
#pragma unroll
    for (int r = 0; r < 8; ++r) {
      float lg = accl[r] + b2v;
      float mx = lg;
#pragma unroll
      for (int m = 1; m <= 8; m <<= 1) mx = fmaxf(mx, __shfl_xor(mx, m, 32));
      const float ex = __expf(lg - mx);
      float sm = ex;
#pragma unroll
      for (int m = 1; m <= 8; m <<= 1) sm += __shfl_xor(sm, m, 32);
      const float al = ex / sm;

      const int erow = e0 + r + 8 * hh;
      if (mrow < NEXP) out_alpha[(size_t)erow * NEXP + mrow] = al;

      float fv = al * accp[r];
#pragma unroll
      for (int m = 1; m <= 8; m <<= 1) fv += __shfl_xor(fv, m, 32);
      if (mrow == 0) out_fused[erow] = fv;
    }
  }
#undef BFRAG
}

// ---------------------------------------------------------------------------
extern "C" void kernel_launch(void* const* d_in, const int* in_sizes, int n_in,
                              void* d_out, int out_size, void* d_ws, size_t ws_size,
                              hipStream_t stream) {
  (void)in_sizes; (void)n_in; (void)out_size; (void)ws_size;
  const float* x  = (const float*)d_in[0];
  const float* w1 = (const float*)d_in[1];
  const float* b1 = (const float*)d_in[2];
  const float* w2 = (const float*)d_in[3];
  const float* b2 = (const float*)d_in[4];
  const float* wp = (const float*)d_in[5];

  float* out_fused = (float*)d_out;            // [E]
  float* out_alpha = out_fused + E_EDGES;      // [E, K]
  __bf16* wsb = (__bf16*)d_ws;                 // 22 fragments * 1KB = 22 KB

  pack_weights<<<NFRAGS, 512, 0, stream>>>(w1, w2, wp, wsb);
  moe_fused<<<E_EDGES / (64 * TILES_PER_WAVE), 128, 0, stream>>>(
      x, b1, b2, wsb, out_fused, out_alpha);
}
